// ACAT_67027259621923
// MI455X (gfx1250) — matrix operand order repaired and checked by reading, not compiled, not run on hardware
//
#include <hip/hip_runtime.h>
#include <hip/hip_bf16.h>
#include <math.h>

// ---------------------------------------------------------------------------
// Problem constants (from the reference)
// ---------------------------------------------------------------------------
#define B_   2
#define H_   8
#define DK_  64
#define L_   2048
#define DH_  512      // H_*DK_
#define SD_  64       // DH_/8
#define LS_  12       // number of strided key positions = ceil(2048/186)
#define MF_  186      // stride (= fq[0])
#define FPADMAX_ 192  // max filter length padded to multiple of 32
#define CONV_TW 128   // t-columns per conv block (4 waves x 2 tiles x 16)
#define WROW 320      // LDS window row stride (>= CONV_TW + 186 - 1 = 313)

typedef __attribute__((ext_vector_type(16))) _Float16 v16h;
typedef __attribute__((ext_vector_type(8)))  _Float16 v8h;
typedef __attribute__((ext_vector_type(8)))  float    v8f;
typedef __attribute__((ext_vector_type(4)))  unsigned int v4u;
typedef __attribute__((ext_vector_type(8)))  int      v8i;
typedef __attribute__((ext_vector_type(4)))  int      v4i;

#if defined(__has_builtin)
#  if __has_builtin(__builtin_amdgcn_tensor_load_to_lds)
#    define HAVE_TDM 1
#  else
#    define HAVE_TDM 0
#  endif
#else
#  define HAVE_TDM 0
#endif

// WMMA operand layouts (CDNA5 ISA):
//  A 16x32 f16 : row M = lane%16; lane group selects K-half;
//                halves 0..7 -> K = base+h, halves 8..15 -> K = 16+base+(h-8).
//  B 32x16 f16 : col N = lane%16; code loads K=0..15 in lanes 0-15, K=16..31 in lanes 16-31 (A too).
//  C/D 16x16 f32: VGPR r -> row r (lanes 0-15) / r+8 (lanes 16-31), col = lane%16.

static __device__ __forceinline__ v16h pack16(v8h lo, v8h hic) {
    v16h r;
#pragma unroll
    for (int h = 0; h < 8; ++h) { r[h] = lo[h]; r[h + 8] = hic[h]; }
    return r;
}

// ---------------------------------------------------------------------------
// init workspace: Qp/Kp = -inf (max-merge identity), BN stats = 0
// ---------------------------------------------------------------------------
__global__ void init_ws(float* __restrict__ Qp, float* __restrict__ Kp,
                        float* __restrict__ stats) {
    int i = blockIdx.x * blockDim.x + threadIdx.x;
    if (i < B_ * L_ * DH_) { Qp[i] = -1e30f; Kp[i] = -1e30f; }
    if (i < 8 * 2 * SD_)   { stats[i] = 0.f; }
}

// ---------------------------------------------------------------------------
// fp32 -> f16 weight conversions (one-time; removes cvt + halves bandwidth on
// the hot weight streams).
// ---------------------------------------------------------------------------
__global__ void cvt_f16(const float* __restrict__ src, _Float16* __restrict__ dst,
                        int n) {
    int i = blockIdx.x * blockDim.x + threadIdx.x;
    if (i < n) dst[i] = (_Float16)src[i];
}

// conv weights (SD,SD,f) -> (SD,SD,fpad) f16, zero-padded in j so every
// 32-wide K-chunk of the implicit GEMM lies inside one ci row.
__global__ void cvt_wpad(const float* __restrict__ src, _Float16* __restrict__ dst,
                         int f, int fpad, int n) {
    int i = blockIdx.x * blockDim.x + threadIdx.x;
    if (i >= n) return;
    int j  = i % fpad;
    int cc = i / fpad;          // cc = co*SD + ci
    dst[i] = (j < f) ? (_Float16)src[cc * f + j] : (_Float16)0.f;
}

// ---------------------------------------------------------------------------
// proj_gemm: C[M,64] = X[M,512] @ Wh[64,512]^T   (M = B*L), Wh pre-cast f16.
// 32 M-rows per block; each wave computes two 16x16 tiles reusing the same
// B operand registers (2 WMMAs per weight load).
// ---------------------------------------------------------------------------
__global__ void __launch_bounds__(128)
proj_gemm(const float* __restrict__ X, const _Float16* __restrict__ Wh,
          _Float16* __restrict__ C) {
    __shared__ _Float16 sA[32 * DH_];       // 32 KB
    const int mt = blockIdx.x * 32;

    for (int idx = threadIdx.x; idx < 32 * DH_; idx += blockDim.x) {
        int r = idx >> 9;
        int k = idx & (DH_ - 1);
        sA[idx] = (_Float16)X[(size_t)(mt + r) * DH_ + k];
    }
    __syncthreads();

    const int wave = threadIdx.x >> 5;
    const int lane = threadIdx.x & 31;
    const int ln   = lane & 15;
    const int hi   = lane >> 4;
    const int nrow = wave * 16 + ln;

    v8f acc0 = {}, acc1 = {};
    for (int k0 = 0; k0 < DH_; k0 += 32) {
        const _Float16* bp = Wh + (size_t)nrow * DH_ + k0 + hi * 16;
        v16h b = pack16(*(const v8h*)bp, *(const v8h*)(bp + 8));
        const _Float16* ap0 = sA + ln * DH_ + k0 + hi * 16;
        v16h a0 = pack16(*(const v8h*)ap0, *(const v8h*)(ap0 + 8));
        const _Float16* ap1 = ap0 + 16 * DH_;
        v16h a1 = pack16(*(const v8h*)ap1, *(const v8h*)(ap1 + 8));
        acc0 = __builtin_amdgcn_wmma_f32_16x16x32_f16(false, a0, false, b,
                                                      (short)0, acc0, false, false);
        acc1 = __builtin_amdgcn_wmma_f32_16x16x32_f16(false, a1, false, b,
                                                      (short)0, acc1, false, false);
    }
#pragma unroll
    for (int r = 0; r < 8; ++r) {
        int row = mt + r + hi * 8;
        int col = wave * 16 + ln;
        C[(size_t)row * SD_ + col]        = (_Float16)acc0[r];
        C[(size_t)(row + 16) * SD_ + col] = (_Float16)acc1[r];
    }
}

// ---------------------------------------------------------------------------
// conv_wmma: implicit-GEMM 1D conv, K ordered as k = ci*fpad + j.
//   Y[b][co][t] = sum_{ci,j} Xc[b][ci][t+j-pad] * W[co][ci][j]
// Block: 16 co x 128 t. Input window (64 ci x 320 t, f16, 40 KB) staged in
// LDS by the Tensor Data Mover when available. Each wave computes two
// adjacent t-tiles with the same weight A-operand (2 WMMAs per weight load,
// halving the dominant L2 weight stream per FLOP).
// BN sum/sumsq accumulated via lane shuffles + global_atomic_add_f32.
// ---------------------------------------------------------------------------
__global__ void __launch_bounds__(128)
conv_wmma(const _Float16* __restrict__ Xc,    // (B, SD, L) f16
          const _Float16* __restrict__ Whp,   // (SD, SD, fpad) f16, zero-padded
          float* __restrict__ Y,              // (B, SD, L)
          float* __restrict__ stat,           // [2][64] : sum, sumsq
          int f, int fpad) {
    __shared__ _Float16 sX[SD_ * WROW];       // 40 KB window
    const int t0  = blockIdx.x * CONV_TW;
    const int co0 = blockIdx.y * 16;
    const int b   = blockIdx.z;
    const int pad = f >> 1;

    const _Float16* Xb = Xc + (size_t)b * SD_ * L_;
#if HAVE_TDM
    {
        int tstart = t0 - pad;
        if (tstart < 0) tstart = 0;
        if (threadIdx.x < 32) {
            unsigned long long gaddr =
                (unsigned long long)(size_t)(Xb + tstart);
            unsigned tdim0 = (unsigned)(L_ - tstart);   // remaining extent
            v4u g0;
            g0[0] = 1u;                                  // count = 1 valid D#
            g0[1] = 0u;                                  // lds_addr: sX at offset 0
            g0[2] = (unsigned)(gaddr & 0xffffffffu);
            g0[3] = (unsigned)((gaddr >> 32) & 0x01ffffffu) | (2u << 30); // type=2
            v8i g1;
            g1[0] = (int)(1u << 16);                     // data_size = 2 bytes
            g1[1] = (int)((tdim0 & 0xffffu) << 16);      // tensor_dim0 lo16
            g1[2] = (int)(((tdim0 >> 16) & 0xffffu) | (64u << 16)); // dim0 hi | dim1
            g1[3] = (int)((unsigned)WROW << 16);         // tile_dim0 = 320
            g1[4] = 64;                                  // tile_dim1 = 64
            g1[5] = L_;                                  // tensor_dim0_stride
            g1[6] = 0;
            g1[7] = 0;
            v4i gz = {0, 0, 0, 0};
#if defined(__clang_major__) && (__clang_major__ >= 23)
            v8i gz8 = {0, 0, 0, 0, 0, 0, 0, 0};
            __builtin_amdgcn_tensor_load_to_lds(g0, g1, gz, gz, gz8, 0);
#else
            __builtin_amdgcn_tensor_load_to_lds(g0, g1, gz, gz, 0);
#endif
            __builtin_amdgcn_s_wait_tensorcnt(0);
        }
        __syncthreads();
    }
#else
    {
        const int Wwin = CONV_TW + f - 1;
        for (int idx = threadIdx.x; idx < SD_ * WROW; idx += blockDim.x) {
            int ci = idx / WROW;
            int tt = idx % WROW;
            int tg = t0 - pad + tt;
            _Float16 v = (_Float16)0.f;
            if (tt < Wwin && tg >= 0 && tg < L_) v = Xb[(size_t)ci * L_ + tg];
            sX[idx] = v;
        }
        __syncthreads();
    }
#endif

    const int wave = threadIdx.x >> 5;
    const int lane = threadIdx.x & 31;
    const int ln   = lane & 15;
    const int hi   = lane >> 4;
    const int tw   = wave * 32;               // two 16-wide t tiles per wave

    v8f acc0 = {}, acc1 = {};
    for (int ci = 0; ci < SD_; ++ci) {
        const _Float16* wrow = Whp + ((size_t)(co0 + ln) * SD_ + ci) * fpad;
        const _Float16* xwin = sX + ci * WROW + tw + ln;
        if (lane == 0) __builtin_prefetch(wrow + fpad, 0, 1);
        for (int j0 = 0; j0 < fpad; j0 += 32) {
            v16h a = pack16(*(const v8h*)(wrow + j0 + hi * 16),
                            *(const v8h*)(wrow + j0 + 8 + hi * 16));
            v16h bv0, bv1;
#pragma unroll
            for (int h = 0; h < 16; ++h) {
                bv0[h] = xwin[j0 + h + hi * 16];
                bv1[h] = xwin[16 + j0 + h + hi * 16];
            }
            acc0 = __builtin_amdgcn_wmma_f32_16x16x32_f16(false, a, false, bv0,
                                                          (short)0, acc0, false, false);
            acc1 = __builtin_amdgcn_wmma_f32_16x16x32_f16(false, a, false, bv1,
                                                          (short)0, acc1, false, false);
        }
    }

#pragma unroll
    for (int r = 0; r < 8; ++r) {
        int co = co0 + r + hi * 8;
        int t  = t0 + tw + ln;
        float v0 = acc0[r];
        float v1 = acc1[r];
        Y[(size_t)b * SD_ * L_ + (size_t)co * L_ + t]      = v0;
        Y[(size_t)b * SD_ * L_ + (size_t)co * L_ + t + 16] = v1;
        float s = v0 + v1, q = v0 * v0 + v1 * v1;
#pragma unroll
        for (int off = 8; off >= 1; off >>= 1) {
            s += __shfl_xor(s, off, 16);
            q += __shfl_xor(q, off, 16);
        }
        if (ln == 0) {
            atomicAdd(&stat[co], s);
            atomicAdd(&stat[SD_ + co], q);
        }
    }
}

// ---------------------------------------------------------------------------
// bn_elu: normalize per channel over (b,t), ELU, convert to f16
// ---------------------------------------------------------------------------
__global__ void bn_elu(const float* __restrict__ Y, const float* __restrict__ stat,
                       const float* __restrict__ gamma, const float* __restrict__ beta,
                       _Float16* __restrict__ Yh) {
    int i = blockIdx.x * blockDim.x + threadIdx.x;
    if (i >= B_ * SD_ * L_) return;
    int c = (i / L_) & (SD_ - 1);
    const float n = (float)(B_ * L_);
    float mean = stat[c] / n;
    float var  = stat[SD_ + c] / n - mean * mean;
    float rs   = rsqrtf(var + 1e-5f);
    float v    = (Y[i] - mean) * rs * gamma[c] + beta[c];
    v = (v > 0.f) ? v : (__expf(v) - 1.f);
    Yh[i] = (_Float16)v;
}

// ---------------------------------------------------------------------------
// branch_proj_max: P[M,512] = max(P, Yh[M,64] @ Wbh[512,64]^T)   (M = B*L)
// 32 M-rows per block; two tiles per wave share the B operand registers.
// ---------------------------------------------------------------------------
__global__ void __launch_bounds__(128)
branch_proj_max(const _Float16* __restrict__ Yh, const _Float16* __restrict__ Wbh,
                float* __restrict__ P) {
    __shared__ _Float16 sA[32 * SD_];
    const int mt = blockIdx.x * 32;
    const int nt = blockIdx.y * 64;

    for (int idx = threadIdx.x; idx < 32 * SD_; idx += blockDim.x)
        sA[idx] = Yh[(size_t)mt * SD_ + idx];
    __syncthreads();

    const int wave = threadIdx.x >> 5;
    const int lane = threadIdx.x & 31;
    const int ln   = lane & 15;
    const int hi   = lane >> 4;
    const int nrow = nt + wave * 16 + ln;

    v8f acc0 = {}, acc1 = {};
#pragma unroll
    for (int k0 = 0; k0 < SD_; k0 += 32) {
        const _Float16* bp = Wbh + (size_t)nrow * SD_ + k0 + hi * 16;
        v16h b = pack16(*(const v8h*)bp, *(const v8h*)(bp + 8));
        const _Float16* ap0 = sA + ln * SD_ + k0 + hi * 16;
        v16h a0 = pack16(*(const v8h*)ap0, *(const v8h*)(ap0 + 8));
        const _Float16* ap1 = ap0 + 16 * SD_;
        v16h a1 = pack16(*(const v8h*)ap1, *(const v8h*)(ap1 + 8));
        acc0 = __builtin_amdgcn_wmma_f32_16x16x32_f16(false, a0, false, b,
                                                      (short)0, acc0, false, false);
        acc1 = __builtin_amdgcn_wmma_f32_16x16x32_f16(false, a1, false, b,
                                                      (short)0, acc1, false, false);
    }
#pragma unroll
    for (int r = 0; r < 8; ++r) {
        int row = mt + r + hi * 8;
        size_t o0 = (size_t)row * DH_ + nrow;
        size_t o1 = (size_t)(row + 16) * DH_ + nrow;
        P[o0] = fmaxf(P[o0], acc0[r]);
        P[o1] = fmaxf(P[o1], acc1[r]);
    }
}

// ---------------------------------------------------------------------------
// vsum: Vs[b,h,d] = sum_k V[b,h,k,d]
// ---------------------------------------------------------------------------
__global__ void vsum_kernel(const float* __restrict__ V, float* __restrict__ Vs) {
    int bh = blockIdx.x;
    int d  = threadIdx.x;
    const float* Vp = V + (size_t)bh * L_ * DK_ + d;
    float s = 0.f;
    for (int k = 0; k < L_; ++k) s += Vp[(size_t)k * DK_];
    Vs[bh * DK_ + d] = s;
}

// ---------------------------------------------------------------------------
// attn_scores: scores on the 12 strided keys, mask, softmax -> attn output
// ---------------------------------------------------------------------------
__global__ void attn_scores(const float* __restrict__ Qp, const float* __restrict__ Kp,
                            const unsigned char* __restrict__ mask,
                            float* __restrict__ attn) {
    int idx = blockIdx.x * blockDim.x + threadIdx.x;   // over B*H*L
    if (idx >= B_ * H_ * L_) return;
    int q  = idx % L_;
    int bh = idx / L_;
    const float* Qr = Qp + ((size_t)bh * L_ + q) * DK_;

    float sc[LS_];
    float mx = -1e30f;
#pragma unroll
    for (int i = 0; i < LS_; ++i) {
        const float* Kr = Kp + ((size_t)bh * L_ + i * MF_) * DK_;
        float d = 0.f;
        for (int e = 0; e < DK_; ++e) d += Qr[e] * Kr[e];
        d *= 0.125f;                                   // 1/sqrt(64)
        if (mask[((size_t)bh * L_ + q) * L_ + i * MF_]) d = -1e9f;
        sc[i] = d;
        mx = fmaxf(mx, d);
    }
    float den = 0.f;
#pragma unroll
    for (int i = 0; i < LS_; ++i) { sc[i] = __expf(sc[i] - mx); den += sc[i]; }
    float inv = 1.f / den;
#pragma unroll
    for (int i = 0; i < LS_; ++i)
        attn[(size_t)idx * LS_ + i] = sc[i] * inv;
}

// ---------------------------------------------------------------------------
// context: algebraic form of softmax(scatter(attn)) @ V
//   ctx[q,d] = (sumV[d] + sum_i (exp(p_i)-1)*V[k_i,d]) / ((L-12) + sum_i exp(p_i))
// ---------------------------------------------------------------------------
__global__ void context_kernel(const float* __restrict__ attn,
                               const float* __restrict__ V,
                               const float* __restrict__ Vs,
                               float* __restrict__ ctx) {
    int idx = blockIdx.x * blockDim.x + threadIdx.x;   // over B*H*L*DK
    if (idx >= B_ * H_ * L_ * DK_) return;
    int d  = idx & (DK_ - 1);
    int q  = (idx >> 6) % L_;
    int bh = idx / (L_ * DK_);
    const float* p = attn + ((size_t)bh * L_ + q) * LS_;
    float den = (float)(L_ - LS_);
    float acc = Vs[bh * DK_ + d];
#pragma unroll
    for (int i = 0; i < LS_; ++i) {
        float w = __expf(p[i]);
        den += w;
        acc += (w - 1.f) * V[((size_t)bh * L_ + i * MF_) * DK_ + d];
    }
    ctx[idx] = acc / den;
}

// ---------------------------------------------------------------------------
// launch
// ---------------------------------------------------------------------------
extern "C" void kernel_launch(void* const* d_in, const int* in_sizes, int n_in,
                              void* d_out, int out_size, void* d_ws, size_t ws_size,
                              hipStream_t stream) {
    (void)in_sizes; (void)n_in; (void)out_size; (void)ws_size;

    const float* Q     = (const float*)d_in[0];
    const float* K     = (const float*)d_in[1];
    const float* V     = (const float*)d_in[2];
    const unsigned char* mask = (const unsigned char*)d_in[3];
    const float* Wq    = (const float*)d_in[4];
    const float* Wk    = (const float*)d_in[5];
    const float* Wbq   = (const float*)d_in[6];
    const float* Wbk   = (const float*)d_in[7];
    const float* gamma = (const float*)d_in[8];
    const float* beta  = (const float*)d_in[9];
    const float* cq[4] = {(const float*)d_in[10], (const float*)d_in[12],
                          (const float*)d_in[14], (const float*)d_in[16]};
    const float* ck[4] = {(const float*)d_in[11], (const float*)d_in[13],
                          (const float*)d_in[15], (const float*)d_in[17]};
    const int F[4]    = {186, 93, 46, 23};
    const int FP[4]   = {192, 96, 64, 32};   // padded to multiple of 32

    // workspace carve-up (f16 regions first, then f32)
    _Float16* Qc   = (_Float16*)d_ws;                         // B*SD*L
    _Float16* Kc   = Qc   + (size_t)B_ * SD_ * L_;
    _Float16* Wqh  = Kc   + (size_t)B_ * SD_ * L_;            // 64*512
    _Float16* Wkh  = Wqh  + (size_t)SD_ * DH_;
    _Float16* Wbqh = Wkh  + (size_t)SD_ * DH_;                // 512*64
    _Float16* Wbkh = Wbqh + (size_t)DH_ * SD_;
    _Float16* Whp  = Wbkh + (size_t)DH_ * SD_;                // 64*64*FPADMAX_
    _Float16* ybf  = Whp  + (size_t)SD_ * SD_ * FPADMAX_;     // B*SD*L
    float*    ybuf = (float*)(ybf + (size_t)B_ * SD_ * L_);   // B*SD*L
    float*    Qp   = ybuf + (size_t)B_ * SD_ * L_;            // B*L*DH
    float*    Kp   = Qp   + (size_t)B_ * L_ * DH_;
    float*    stats= Kp   + (size_t)B_ * L_ * DH_;            // 8 slots x 128
    float*    Vsum = stats + 8 * 2 * SD_;

    float* ctx_out  = (float*)d_out;
    float* attn_out = ctx_out + (size_t)B_ * H_ * L_ * DK_;

    // 0) init + one-time weight conversions
    init_ws<<<(B_ * L_ * DH_ + 255) / 256, 256, 0, stream>>>(Qp, Kp, stats);
    cvt_f16<<<(SD_ * DH_ + 255) / 256, 256, 0, stream>>>(Wq,  Wqh,  SD_ * DH_);
    cvt_f16<<<(SD_ * DH_ + 255) / 256, 256, 0, stream>>>(Wk,  Wkh,  SD_ * DH_);
    cvt_f16<<<(DH_ * SD_ + 255) / 256, 256, 0, stream>>>(Wbq, Wbqh, DH_ * SD_);
    cvt_f16<<<(DH_ * SD_ + 255) / 256, 256, 0, stream>>>(Wbk, Wbkh, DH_ * SD_);

    // 1) channel projections (WMMA)
    proj_gemm<<<(B_ * L_) / 32, 128, 0, stream>>>(Q, Wqh, Qc);
    proj_gemm<<<(B_ * L_) / 32, 128, 0, stream>>>(K, Wkh, Kc);

    // 2) 8 conv branches: conv (WMMA implicit GEMM, TDM-staged) -> BN+ELU -> proj+max
    dim3 cgrid(L_ / CONV_TW, SD_ / 16, B_);
    dim3 pgrid((B_ * L_) / 32, DH_ / 64, 1);
    for (int i = 0; i < 4; ++i) {
        int npw = SD_ * SD_ * FP[i];
        // Q branch i (stats slot i)
        cvt_wpad<<<(npw + 255) / 256, 256, 0, stream>>>(cq[i], Whp, F[i], FP[i], npw);
        conv_wmma<<<cgrid, 128, 0, stream>>>(Qc, Whp, ybuf,
                                             stats + (size_t)i * 128, F[i], FP[i]);
        bn_elu<<<(B_ * SD_ * L_) / 256, 256, 0, stream>>>(ybuf, stats + (size_t)i * 128,
                                                          gamma, beta, ybf);
        branch_proj_max<<<pgrid, 128, 0, stream>>>(ybf, Wbqh, Qp);
        // K branch i (stats slot 4+i)
        cvt_wpad<<<(npw + 255) / 256, 256, 0, stream>>>(ck[i], Whp, F[i], FP[i], npw);
        conv_wmma<<<cgrid, 128, 0, stream>>>(Kc, Whp, ybuf,
                                             stats + (size_t)(4 + i) * 128, F[i], FP[i]);
        bn_elu<<<(B_ * SD_ * L_) / 256, 256, 0, stream>>>(ybuf, stats + (size_t)(4 + i) * 128,
                                                          gamma, beta, ybf);
        branch_proj_max<<<pgrid, 128, 0, stream>>>(ybf, Wbkh, Kp);
    }

    // 3) attention on strided keys + algebraic second softmax
    vsum_kernel<<<B_ * H_, DK_, 0, stream>>>(V, Vsum);
    attn_scores<<<(B_ * H_ * L_) / 256, 256, 0, stream>>>(Qp, Kp, mask, attn_out);
    context_kernel<<<(B_ * H_ * L_ * DK_) / 256, 256, 0, stream>>>(attn_out, V, Vsum, ctx_out);
}
